// EdgeTransformerEncoder_new_13348758355983
// MI455X (gfx1250) — matrix operand order repaired and checked by reading, not compiled, not run on hardware
//
#include <hip/hip_runtime.h>
#include <hip/hip_bf16.h>
#include <math.h>

// ======================================================================
// EdgeTransformerEncoder for MI455X (gfx1250, wave32, WMMA + TDM)
//
// Dense lins: TDM (tensor_load_to_lds) stages A(16xK) and B(Kx32) f32
// tiles into LDS, one wave issues the DMA, s_wait_tensorcnt + barrier,
// then a fully-unrolled chain of v_wmma_f32_16x16x32_f16 (f32 demoted
// to f16 fragments, f32 accumulate). Edge softmax/aggregation: atomic
// kernels. BatchNorm stats: LDS atomics. Edge-feature projections are
// recomputed on the fly (saves ~410MB of workspace).
// ======================================================================

#define NN     20000
#define EE     200000
#define NHEADS 4
static constexpr float EPSBN = 1e-5f;

typedef __attribute__((ext_vector_type(16))) _Float16 v16h;
typedef __attribute__((ext_vector_type(8)))  float    v8f;
typedef __attribute__((ext_vector_type(4)))  unsigned v4u;
typedef __attribute__((ext_vector_type(8)))  int      v8i;
typedef __attribute__((ext_vector_type(4)))  int      v4i;

// ---------------- small utility kernels ----------------
__global__ void k_zero(float* __restrict__ p, long n) {
  long i  = (long)blockIdx.x * blockDim.x + threadIdx.x;
  long st = (long)gridDim.x * blockDim.x;
  for (; i < n; i += st) p[i] = 0.f;
}

__global__ void k_deg(const int* __restrict__ ei, float* __restrict__ deg, int E) {
  int e = blockIdx.x * blockDim.x + threadIdx.x;
  if (e < E) atomicAdd(&deg[ei[E + e]], 1.0f);
}

__global__ void k_norm(const int* __restrict__ ei, const float* __restrict__ deg,
                       float* __restrict__ nrm, int E) {
  int e = blockIdx.x * blockDim.x + threadIdx.x;
  if (e >= E) return;
  float a = rsqrtf(fmaxf(deg[ei[e]], 1.f));
  float b = rsqrtf(fmaxf(deg[ei[E + e]], 1.f));
  nrm[e] = a * b;
}

// out[dst,c] += h[src,c] * norm[e]
__global__ void k_scatter(const int* __restrict__ ei, const float* __restrict__ h,
                          const float* __restrict__ nrm, float* __restrict__ out,
                          int E, int C) {
  long t = (long)blockIdx.x * blockDim.x + threadIdx.x;
  if (t >= (long)E * C) return;
  int e = (int)(t / C), c = (int)(t % C);
  atomicAdd(&out[(size_t)ei[E + e] * C + c], h[(size_t)ei[e] * C + c] * nrm[e]);
}

__global__ void k_add(const float* __restrict__ a, const float* __restrict__ b,
                      float* __restrict__ y, long n) {
  long t = (long)blockIdx.x * blockDim.x + threadIdx.x;
  if (t < n) y[t] = a[t] + b[t];
}

// ---------------- TDM: 2-D tile DMA into LDS (D# per ISA 08 §8.3/8.4) ---
// data_size = 4 bytes (code 2). Groups 2/3 zero (<=2-D tensor).
__device__ __forceinline__ void tdm_load_2d(unsigned lds_byte_addr, const void* gptr,
                                            unsigned tensor_d0, unsigned tensor_d1,
                                            unsigned tile_d0, unsigned tile_d1,
                                            unsigned stride_d0_elems) {
  unsigned long long ga = (unsigned long long)(uintptr_t)gptr;
  v4u g0;
  g0[0] = 1u;                                              // count=1, user desc
  g0[1] = lds_byte_addr;                                   // lds_addr [63:32]
  g0[2] = (unsigned)(ga & 0xFFFFFFFFull);                  // global_addr[31:0]
  g0[3] = (unsigned)((ga >> 32) & 0x1FFFFFFull)            // global_addr[56:32]
          | (2u << 30);                                    // type = 2 ("image")
  v8i g1;
  g1[0] = (int)(2u << 16);                                 // data_size = 4B
  g1[1] = (int)((tensor_d0 & 0xFFFFu) << 16);              // tensor_dim0 lo16 @[63:48]
  g1[2] = (int)((tensor_d0 >> 16) | ((tensor_d1 & 0xFFFFu) << 16)); // d0 hi | d1 lo
  g1[3] = (int)((tensor_d1 >> 16) | (tile_d0 << 16));      // d1 hi | tile_dim0
  g1[4] = (int)(tile_d1 & 0xFFFFu);                        // tile_dim1 (tile_dim2=0)
  g1[5] = (int)stride_d0_elems;                            // tensor_dim0_stride[31:0]
  g1[6] = 0;                                               // stride hi | dim1_stride lo
  g1[7] = 0;
  v4i z4 = {};
  v8i z8 = {};
  __builtin_amdgcn_tensor_load_to_lds(g0, g1, z4, z4, z8, 0);
}

// ---------------- WMMA GEMM: Y[rows,cols] = X[rows,K] @ W[K,cols] + b ----
// rows%16==0, cols%32==0, K in {32,64,96,128}. Block = 64 threads (2 waves),
// output tile 16 rows x 32 cols (wave w owns 16 cols). TDM stages both
// operand tiles in LDS; K is compile-time -> unrolled WMMA chain.
template <int K>
__global__ void k_gemm_wmma(const float* __restrict__ X, const float* __restrict__ W,
                            const float* __restrict__ bias, float* __restrict__ Y,
                            int cols) {
  __shared__ float lA[16 * K];   // A tile, row-major [16][K]
  __shared__ float lB[K * 32];   // B tile, row-major [K][32]

  const int tid  = threadIdx.x;        // 0..63
  const int w    = tid >> 5;           // wave id (uniform per wave)
  const int lane = tid & 31;
  const int grp  = lane >> 4;
  const int idx  = lane & 15;

  if (tid < 32) {  // wave-uniform branch: wave 0 issues both DMAs
    tdm_load_2d((unsigned)(uintptr_t)lA,
                X + (size_t)blockIdx.x * 16 * K,
                /*tensor_d0=*/K, /*tensor_d1=*/16,
                /*tile_d0=*/K, /*tile_d1=*/16, /*stride=*/K);
    tdm_load_2d((unsigned)(uintptr_t)lB,
                W + (size_t)blockIdx.y * 32,
                /*tensor_d0=*/(unsigned)cols, /*tensor_d1=*/K,
                /*tile_d0=*/32, /*tile_d1=*/K, /*stride=*/(unsigned)cols);
    __builtin_amdgcn_s_wait_tensorcnt(0);
  }
  __syncthreads();

  v8f acc = {};
#pragma unroll
  for (int k0 = 0; k0 < K; k0 += 32) {
    // A fragment: lane row = idx, K = k0 + 16*grp + j
    const float* ar = lA + idx * K + k0 + grp * 16;
    // B fragment: K = k0 + 16*grp + j, column = w*16 + idx
    const float* br = lB + (k0 + grp * 16) * 32 + (w * 16 + idx);
    v16h a, b;
#pragma unroll
    for (int j = 0; j < 8; ++j) {
      a[j]     = (_Float16)ar[j];
      a[j + 8] = (_Float16)ar[8 + j];
    }
#pragma unroll
    for (int j = 0; j < 16; ++j)
      b[j] = (_Float16)br[j * 32];
    acc = __builtin_amdgcn_wmma_f32_16x16x32_f16(
        false, a, false, b, (short)0, acc, false, false);
  }

  // C/D layout: vgpr i -> row = 16*tileM + i + 8*grp, col = 32*tileN + w*16 + idx
  const int col = blockIdx.y * 32 + w * 16 + idx;
  const float bc = bias[col];
  float* yp = Y + (size_t)(blockIdx.x * 16 + grp * 8) * cols + col;
#pragma unroll
  for (int i = 0; i < 8; ++i) yp[(size_t)i * cols] = acc[i] + bc;
}

// ---------------- BatchNorm (C = 64 fixed) ----------------
__global__ void k_bnstats(const float* __restrict__ x, float* __restrict__ gs,
                          float* __restrict__ gq, int n) {
  __shared__ float ls[64], lq[64];
  for (int c = threadIdx.x; c < 64; c += blockDim.x) { ls[c] = 0.f; lq[c] = 0.f; }
  __syncthreads();
  int row = blockIdx.x * blockDim.x + threadIdx.x;
  if (row < n) {
    for (int c = 0; c < 64; ++c) {
      float v = x[(size_t)row * 64 + c];
      atomicAdd(&ls[c], v);        // ds_add_f32
      atomicAdd(&lq[c], v * v);
    }
  }
  __syncthreads();
  for (int c = threadIdx.x; c < 64; c += blockDim.x) {
    atomicAdd(&gs[c], ls[c]);
    atomicAdd(&gq[c], lq[c]);
  }
}

__global__ void k_bnapply(float* __restrict__ x, const float* __restrict__ gs,
                          const float* __restrict__ gq, const float* __restrict__ gamma,
                          const float* __restrict__ beta, int n) {
  long t = (long)blockIdx.x * blockDim.x + threadIdx.x;
  if (t >= (long)n * 64) return;
  int c = (int)(t % 64);
  float mu  = gs[c] / (float)n;
  float var = gq[c] / (float)n - mu * mu;
  x[t] = (x[t] - mu) * rsqrtf(var + EPSBN) * gamma[c] + beta[c];
}

// ---------------- monotone float<->uint for atomic segment-max ----------
__device__ __forceinline__ unsigned fenc(float f) {
  unsigned u = __float_as_uint(f);
  return (u & 0x80000000u) ? ~u : (u | 0x80000000u);
}
__device__ __forceinline__ float fdec(unsigned u) {
  return (u & 0x80000000u) ? __uint_as_float(u & 0x7FFFFFFFu) : __uint_as_float(~u);
}

// On-the-fly edge projection: (edge_attr[e] @ We + be)[hc]
__device__ __forceinline__ float eproj(const float* __restrict__ ea,
                                       const float* __restrict__ We,
                                       const float* __restrict__ be,
                                       int e, int edim, int HD, int hc) {
  float acc = be[hc];
  const float* a = ea + (size_t)e * edim;
  for (int d = 0; d < edim; ++d) acc = fmaf(a[d], We[(size_t)d * HD + hc], acc);
  return acc;
}

// logits[e,h] = <q[dst,h,:], k[src,h,:] (+e)> / sqrt(cph); segment max into m
__global__ void k_logits(const int* __restrict__ ei, const float* __restrict__ q,
                         const float* __restrict__ k, const float* __restrict__ ea,
                         const float* __restrict__ We, const float* __restrict__ be,
                         int edim, float* __restrict__ logits, unsigned* __restrict__ m,
                         int E, int cph, int HD, float rscale) {
  int t = blockIdx.x * blockDim.x + threadIdx.x;
  if (t >= E * NHEADS) return;
  int e = t / NHEADS, h = t % NHEADS;
  int s = ei[e], d = ei[E + e];
  const float* qd = q + (size_t)d * HD + h * cph;
  const float* ks = k + (size_t)s * HD + h * cph;
  float acc = 0.f;
  for (int c = 0; c < cph; ++c) {
    float kk = ks[c];
    if (We) kk += eproj(ea, We, be, e, edim, HD, h * cph + c);
    acc = fmaf(qd[c], kk, acc);
  }
  float lg = acc * rscale;
  logits[t] = lg;
  atomicMax(&m[(size_t)d * NHEADS + h], fenc(lg));
}

__global__ void k_expsum(const int* __restrict__ ei, const float* __restrict__ logits,
                         const unsigned* __restrict__ m, float* __restrict__ exb,
                         float* __restrict__ sden, int E) {
  int t = blockIdx.x * blockDim.x + threadIdx.x;
  if (t >= E * NHEADS) return;
  int e = t / NHEADS, h = t % NHEADS;
  int d = ei[E + e];
  float mv = fdec(m[(size_t)d * NHEADS + h]);
  if (!__builtin_isfinite(mv)) mv = 0.f;
  float ex = __expf(logits[t] - mv);
  exb[t] = ex;
  atomicAdd(&sden[(size_t)d * NHEADS + h], ex);
}

// alpha = ex / sum; agg[dst,h,c] += alpha * (v[src,h,c] (+e))
__global__ void k_agg(const int* __restrict__ ei, const float* __restrict__ exb,
                      const float* __restrict__ sden, const float* __restrict__ v,
                      const float* __restrict__ ea, const float* __restrict__ We,
                      const float* __restrict__ be, int edim,
                      float* __restrict__ alphaOut, float* __restrict__ agg,
                      int E, int cph, int HD) {
  int t = blockIdx.x * blockDim.x + threadIdx.x;
  if (t >= E * NHEADS) return;
  int e = t / NHEADS, h = t % NHEADS;
  int s = ei[e], d = ei[E + e];
  float al = exb[t] / (sden[(size_t)d * NHEADS + h] + 1e-16f);
  alphaOut[t] = al;
  const float* vs = v + (size_t)s * HD + h * cph;
  float* ag = agg + (size_t)d * HD + h * cph;
  for (int c = 0; c < cph; ++c) {
    float vv = vs[c];
    if (We) vv += eproj(ea, We, be, e, edim, HD, h * cph + c);
    atomicAdd(&ag[c], al * vv);
  }
}

__global__ void k_headmean(const float* __restrict__ agg, float* __restrict__ out,
                           int n, int cph, int HD) {
  long t = (long)blockIdx.x * blockDim.x + threadIdx.x;
  if (t >= (long)n * cph) return;
  int i = (int)(t / cph), c = (int)(t % cph);
  float s = 0.f;
  for (int h = 0; h < NHEADS; ++h) s += agg[(size_t)i * HD + h * cph + c];
  out[t] = s * (1.0f / NHEADS);
}

// beta gating: g = sigmoid([skip,out,skip-out] @ Wb + bb); y = g*skip+(1-g)*out
__global__ void k_betagate(const float* __restrict__ skip, const float* __restrict__ out,
                           const float* __restrict__ Wb, const float* __restrict__ bb,
                           float* __restrict__ y, int n, int dout) {
  int i = blockIdx.x * blockDim.x + threadIdx.x;
  if (i >= n) return;
  const float* sk = skip + (size_t)i * dout;
  const float* ot = out  + (size_t)i * dout;
  float z = bb[0];
  for (int j = 0; j < dout; ++j)
    z += sk[j] * Wb[j] + ot[j] * Wb[dout + j] + (sk[j] - ot[j]) * Wb[2 * dout + j];
  float g = 1.f / (1.f + __expf(-z));
  float* yy = y + (size_t)i * dout;
  for (int j = 0; j < dout; ++j) yy[j] = g * sk[j] + (1.f - g) * ot[j];
}

// ======================================================================
extern "C" void kernel_launch(void* const* d_in, const int* in_sizes, int n_in,
                              void* d_out, int out_size, void* d_ws, size_t ws_size,
                              hipStream_t stream) {
  (void)in_sizes; (void)n_in; (void)out_size; (void)ws_size;

  // ---- inputs (params flattened in Python dict insertion order) ----
  const float* x  = (const float*)d_in[0];
  const int*   ei = (const int*)d_in[1];
  const float* ea = (const float*)d_in[2];
  auto F = [&](int i) { return (const float*)d_in[i]; };
  // 3,4: conv1 W,b | 5,6: bn gamma,beta | 7,8: conem W,b
  // conv2 q,k,v,skip,e,beta(W,b each): 9..20
  // layers[0]: 21..32 | layers[1]: 33..44 | mu: 45..52 | logstd: 53..60

  // ---- workspace carving ----
  float* W0 = (float*)d_ws;
  size_t o = 0;
  auto carve = [&](size_t nf) { float* p = W0 + o; o += nf; return p; };
  float* deg = carve(NN);
  float* nrm = carve(EE);
  float* tmp = carve((size_t)NN * 64);
  float* x1  = carve((size_t)NN * 64);
  float* x2  = carve((size_t)NN * 32);
  float* qb  = carve((size_t)NN * 512);
  float* kb  = carve((size_t)NN * 512);
  float* vb  = carve((size_t)NN * 512);
  float* lg  = carve((size_t)EE * NHEADS);
  float* exb = carve((size_t)EE * NHEADS);
  float* alA = carve((size_t)EE * NHEADS);
  float* alB = carve((size_t)EE * NHEADS);
  float* mb  = carve((size_t)NN * NHEADS);
  float* sd  = carve((size_t)NN * NHEADS);
  float* agg = carve((size_t)NN * 512);
  float* skb = carve((size_t)NN * 128);
  float* omn = carve((size_t)NN * 128);
  float* x3  = carve((size_t)NN * 128);
  float* x4  = carve((size_t)NN * 96);
  float* x5  = carve((size_t)NN * 64);
  float* bns = carve(64);
  float* bnq = carve(64);

  auto Z = [&](float* p, long n) {
    long b = (n + 255) / 256; if (b > 4096) b = 4096;
    k_zero<<<(int)b, 256, 0, stream>>>(p, n);
  };
  auto GEMM = [&](const float* X, const float* Wt, const float* b, float* Y,
                  int rows, int K, int cols) {
    dim3 g(rows / 16, cols / 32);
    switch (K) {
      case 32:  k_gemm_wmma<32> <<<g, 64, 0, stream>>>(X, Wt, b, Y, cols); break;
      case 64:  k_gemm_wmma<64> <<<g, 64, 0, stream>>>(X, Wt, b, Y, cols); break;
      case 96:  k_gemm_wmma<96> <<<g, 64, 0, stream>>>(X, Wt, b, Y, cols); break;
      case 128: k_gemm_wmma<128><<<g, 64, 0, stream>>>(X, Wt, b, Y, cols); break;
      default: break;  // all call sites use K in {32,64,96,128}
    }
  };

  auto TCONV = [&](const float* xin, int din, int cph, bool concat, bool hasBeta,
                   const float* Wq, const float* bq, const float* Wk, const float* bk_,
                   const float* Wv, const float* bv, const float* Wsk, const float* bsk,
                   const float* We, const float* be, int edim, const float* eattr,
                   const float* Wb, const float* bb, float* alphaOut, float* xout) {
    int HD = NHEADS * cph;
    GEMM(xin, Wq, bq, qb, NN, din, HD);
    GEMM(xin, Wk, bk_, kb, NN, din, HD);
    GEMM(xin, Wv, bv, vb, NN, din, HD);
    Z(mb, (long)NN * NHEADS);
    Z(sd, (long)NN * NHEADS);
    Z(agg, (long)NN * HD);
    int eh  = EE * NHEADS;
    int blk = (eh + 255) / 256;
    k_logits<<<blk, 256, 0, stream>>>(ei, qb, kb, eattr, We, be, edim, lg,
                                      (unsigned*)mb, EE, cph, HD, rsqrtf((float)cph));
    k_expsum<<<blk, 256, 0, stream>>>(ei, lg, (const unsigned*)mb, exb, sd, EE);
    k_agg<<<blk, 256, 0, stream>>>(ei, exb, sd, vb, eattr, We, be, edim,
                                   alphaOut, agg, EE, cph, HD);
    int dout = concat ? HD : cph;
    GEMM(xin, Wsk, bsk, skb, NN, din, dout);
    const float* outsrc;
    if (!concat) {
      long t = (long)NN * cph;
      k_headmean<<<(int)((t + 255) / 256), 256, 0, stream>>>(agg, omn, NN, cph, HD);
      outsrc = omn;
    } else {
      outsrc = agg;
    }
    if (hasBeta) {
      k_betagate<<<(NN + 255) / 256, 256, 0, stream>>>(skb, outsrc, Wb, bb, xout, NN, dout);
    } else {
      long t = (long)NN * dout;
      k_add<<<(int)((t + 255) / 256), 256, 0, stream>>>(outsrc, skb, xout, t);
    }
  };

  // ---- GCN stage 1: x@W1+b1 -> scatter(norm) -> BN ----
  Z(deg, NN);
  k_deg<<<(EE + 255) / 256, 256, 0, stream>>>(ei, deg, EE);
  k_norm<<<(EE + 255) / 256, 256, 0, stream>>>(ei, deg, nrm, EE);
  GEMM(x, F(3), F(4), tmp, NN, 128, 64);
  Z(x1, (long)NN * 64);
  k_scatter<<<(int)(((long)EE * 64 + 255) / 256), 256, 0, stream>>>(ei, tmp, nrm, x1, EE, 64);
  Z(bns, 64); Z(bnq, 64);
  k_bnstats<<<(NN + 255) / 256, 256, 0, stream>>>(x1, bns, bnq, NN);
  k_bnapply<<<(int)(((long)NN * 64 + 255) / 256), 256, 0, stream>>>(x1, bns, bnq, F(5), F(6), NN);

  // ---- GCN stage 2 (conem) ----
  GEMM(x1, F(7), F(8), tmp, NN, 64, 32);
  Z(x2, (long)NN * 32);
  k_scatter<<<(int)(((long)EE * 32 + 255) / 256), 256, 0, stream>>>(ei, tmp, nrm, x2, EE, 32);

  // ---- conv2: din=32, cph=128, edge_dim=16, beta, mean-over-heads ----
  TCONV(x2, 32, 128, false, true,
        F(9), F(10), F(11), F(12), F(13), F(14), F(15), F(16),
        F(17), F(18), 16, ea, F(19), F(20), alA, x3);

  // ---- layers[0]: din=128, cph=96, edge_dim=4 (prev alpha) ----
  TCONV(x3, 128, 96, false, true,
        F(21), F(22), F(23), F(24), F(25), F(26), F(27), F(28),
        F(29), F(30), 4, alA, F(31), F(32), alB, x4);

  // ---- layers[1]: din=96, cph=64, edge_dim=4 ----
  TCONV(x4, 96, 64, false, true,
        F(33), F(34), F(35), F(36), F(37), F(38), F(39), F(40),
        F(41), F(42), 4, alB, F(43), F(44), alA, x5);

  // ---- mu / logstd: din=64, cph=32, concat, no edge, no beta ----
  float* out = (float*)d_out;
  TCONV(x5, 64, 32, true, false,
        F(45), F(46), F(47), F(48), F(49), F(50), F(51), F(52),
        nullptr, nullptr, 0, nullptr, nullptr, nullptr, alB, out);
  TCONV(x5, 64, 32, true, false,
        F(53), F(54), F(55), F(56), F(57), F(58), F(59), F(60),
        nullptr, nullptr, 0, nullptr, nullptr, nullptr, alB, out + (size_t)NN * 128);
}
